// BiPointNet2SSGCls_35158602285609
// MI455X (gfx1250) — compile-verified, matrix operand order checked by reading, not run on hardware
//
#include <hip/hip_runtime.h>
#include <stdint.h>
#include <stddef.h>

typedef _Float16 h16;
typedef __attribute__((ext_vector_type(16))) _Float16 v16h;
typedef __attribute__((ext_vector_type(8)))  _Float16 v8h;
typedef __attribute__((ext_vector_type(8)))  float    v8f;

#define BN_EPS_F 1e-5f

// ============================================================================
// WMMA binarized GEMM.
//   A : M x Kp row-major f16 signs (+1/-1, 0.0 in padded K columns)
//   Bw: N x Kp row-major f16 signs (row n = binarized weight row / column of B)
//   Out[m,n] = alpha[n] * sum_k A[m,k] * Bw[n,k]
// M, N multiples of 16; Kp multiple of 32. One wave -> one 16x16 tile.
// A-fragment layout (ISA 7.12.2, 16-bit A 16x32): lanes 0-15 hold row m,
// K = {0..7,16..23}; lanes 16-31 hold K = {8..15,24..31}. B mirrored with N on
// lanes. C/D: VGPR r -> row m0+r (+8 for lanes 16-31), col n0+(lane&15).
// ============================================================================
static __device__ __forceinline__ v16h load_frag32(const h16* p) {
  v8h lo = *(const v8h*)(p);
  v8h hi = *(const v8h*)(p + 16);
  v16h r;
#pragma unroll
  for (int i = 0; i < 8; ++i) { r[i] = lo[i]; r[i + 8] = hi[i]; }
  return r;
}

__global__ void gemm_bin16(const h16* __restrict__ A, const h16* __restrict__ Bw,
                           const float* __restrict__ alpha, float* __restrict__ Out,
                           int M, int N, int Kp)
{
  int wave = blockIdx.x * ((int)blockDim.x >> 5) + ((int)threadIdx.x >> 5);
  int lane = (int)threadIdx.x & 31;
  int tilesN = N >> 4;
  int tilesM = M >> 4;
  if (wave >= tilesM * tilesN) return;   // wave-uniform: EXEC stays all-ones
  int tm = wave / tilesN;
  int tn = wave - tm * tilesN;
  int hsel = lane >> 4;                  // 0: lanes 0-15, 1: lanes 16-31
  const h16* ap = A  + (size_t)(tm * 16 + (lane & 15)) * Kp + hsel * 8;
  const h16* bp = Bw + (size_t)(tn * 16 + (lane & 15)) * Kp + hsel * 8;
  v8f acc = {};
  for (int k = 0; k < Kp; k += 32) {
    v16h a = load_frag32(ap + k);
    v16h b = load_frag32(bp + k);
    acc = __builtin_amdgcn_wmma_f32_16x16x32_f16(false, a, false, b,
                                                 (short)0, acc, false, false);
  }
  int col = tn * 16 + (lane & 15);
  float al = alpha[col];
  int rbase = tm * 16 + hsel * 8;
#pragma unroll
  for (int r = 0; r < 8; ++r)
    Out[(size_t)(rbase + r) * N + col] = acc[r] * al;
}

// ============================================================================
// Weight prep: alpha[o] = mean|W[o,:]| ; signW[o,k] = sign(W[o,k]) f16, zero pad
// ============================================================================
__global__ void prep_weights(const float* __restrict__ W, int O, int I,
                             h16* __restrict__ S, float* __restrict__ alpha,
                             int Opad, int Kp)
{
  int o = blockIdx.x;
  __shared__ float red[256];
  float s = 0.f;
  if (o < O)
    for (int i = threadIdx.x; i < I; i += blockDim.x) s += fabsf(W[(size_t)o * I + i]);
  red[threadIdx.x] = s;
  __syncthreads();
  for (int off = 128; off > 0; off >>= 1) {
    if ((int)threadIdx.x < off) red[threadIdx.x] += red[threadIdx.x + off];
    __syncthreads();
  }
  if (threadIdx.x == 0) alpha[o] = (o < O) ? red[0] / (float)I : 0.f;
  for (int i = threadIdx.x; i < Kp; i += blockDim.x) {
    h16 v = (h16)0.0f;
    if (o < O && i < I) v = (W[(size_t)o * I + i] >= 0.f) ? (h16)1.0f : (h16)-1.0f;
    S[(size_t)o * Kp + i] = v;
  }
}

// ============================================================================
// Farthest point sampling: one block per batch; points + dists in LDS.
// Matches jax.lax.scan semantics: idx[0]=0, tie-break = first index (argmax).
// ============================================================================
__global__ void fps_kernel(const float* __restrict__ posg, int B, int N, int S,
                           int* __restrict__ cidx, float* __restrict__ centers)
{
  extern __shared__ float sdyn[];
  float* dists = sdyn;
  float* px = sdyn + N;
  float* py = sdyn + 2 * N;
  float* pz = sdyn + 3 * N;
  __shared__ float rv[256];
  __shared__ int   ri[256];
  int b = blockIdx.x;
  int tid = threadIdx.x, bd = blockDim.x;
  for (int i = tid; i < N; i += bd) {
    const float* p = posg + ((size_t)b * N + i) * 3;
    px[i] = p[0]; py[i] = p[1]; pz[i] = p[2];
    dists[i] = 1e10f;
  }
  __syncthreads();
  int last = 0;
  for (int s = 0; s < S; ++s) {
    if (tid == 0) {
      cidx[b * S + s] = last;
      float* c = centers + ((size_t)b * S + s) * 3;
      c[0] = px[last]; c[1] = py[last]; c[2] = pz[last];
    }
    float lx = px[last], ly = py[last], lz = pz[last];
    float best = -1.f; int bidx = 0x7fffffff;
    for (int i = tid; i < N; i += bd) {
      float dx = px[i] - lx, dy = py[i] - ly, dz = pz[i] - lz;
      float d = dx * dx + dy * dy + dz * dz;
      float nd = fminf(dists[i], d);
      dists[i] = nd;
      if (nd > best) { best = nd; bidx = i; }   // ascending i => first-index tie
    }
    rv[tid] = best; ri[tid] = bidx;
    __syncthreads();
    for (int off = bd >> 1; off > 0; off >>= 1) {
      if (tid < off) {
        float ov = rv[tid + off]; int oi = ri[tid + off];
        if (ov > rv[tid] || (ov == rv[tid] && oi < ri[tid])) { rv[tid] = ov; ri[tid] = oi; }
      }
      __syncthreads();
    }
    last = ri[0];
    __syncthreads();
  }
}

// ============================================================================
// Ball query: one wave32 per center; ordered compaction via ballot/popcount.
// Collects the first K in-radius indices in ascending order, pads with first.
// ============================================================================
__global__ void ball_query_kernel(const float* __restrict__ posg,
                                  const float* __restrict__ centers,
                                  int B, int N, int S, float r2, int Kq,
                                  int* __restrict__ gi)
{
  int wid = blockIdx.x * ((int)blockDim.x >> 5) + ((int)threadIdx.x >> 5);
  int lane = (int)threadIdx.x & 31;
  if (wid >= B * S) return;
  int b = wid / S;
  const float* c = centers + (size_t)wid * 3;
  float cx = c[0], cy = c[1], cz = c[2];
  int* out = gi + (size_t)wid * Kq;
  int collected = 0, first = 0;
  for (int basep = 0; basep < N && collected < Kq; basep += 32) {
    int i = basep + lane;
    bool inside = false;
    if (i < N) {
      const float* p = posg + ((size_t)b * N + i) * 3;
      float dx = p[0] - cx, dy = p[1] - cy, dz = p[2] - cz;
      inside = (dx * dx + dy * dy + dz * dz) <= r2;
    }
    unsigned m = (unsigned)__ballot(inside);
    if (collected == 0 && m) first = basep + __ffs(m) - 1;
    int rank = __popc(m & ((1u << lane) - 1u));
    if (inside && collected + rank < Kq) out[collected + rank] = i;
    collected += __popc(m);
    if (collected > Kq) collected = Kq;
  }
  for (int k = collected + lane; k < Kq; k += 32) out[k] = first;
}

// ============================================================================
// Grouping kernels: build sign-encoded f16 rows (K-padded with 0.0)
// ============================================================================
__device__ __forceinline__ h16 sgn(float v) { return (v >= 0.f) ? (h16)1.0f : (h16)-1.0f; }

__global__ void group1(const float* __restrict__ posg, const float* __restrict__ ctr,
                       const int* __restrict__ gi, int B, int N, int S, int Kq,
                       h16* __restrict__ out, int Kp)
{
  size_t rows = (size_t)B * S * Kq;
  for (size_t r = blockIdx.x * (size_t)blockDim.x + threadIdx.x; r < rows;
       r += (size_t)gridDim.x * blockDim.x) {
    size_t t = r / Kq;
    int j = (int)(t % S);
    int b = (int)(t / S);
    int p = gi[r]; if (p >= N || p < 0) p = 0;
    const float* pp = posg + ((size_t)b * N + p) * 3;
    const float* cc = ctr + ((size_t)b * S + j) * 3;
    h16* o = out + r * Kp;
    o[0] = sgn(pp[0] - cc[0]);
    o[1] = sgn(pp[1] - cc[1]);
    o[2] = sgn(pp[2] - cc[2]);
    for (int c = 3; c < Kp; ++c) o[c] = (h16)0.0f;
  }
}

__global__ void group2(const float* __restrict__ posg, const float* __restrict__ ctr,
                       const int* __restrict__ gi, const float* __restrict__ feat,
                       int B, int Np, int S, int Kq, int C,
                       h16* __restrict__ out, int Kp)
{
  size_t rows = (size_t)B * S * Kq;
  for (size_t r = blockIdx.x * (size_t)blockDim.x + threadIdx.x; r < rows;
       r += (size_t)gridDim.x * blockDim.x) {
    size_t t = r / Kq;
    int j = (int)(t % S);
    int b = (int)(t / S);
    int p = gi[r]; if (p >= Np || p < 0) p = 0;
    const float* pp = posg + ((size_t)b * Np + p) * 3;
    const float* cc = ctr + ((size_t)b * S + j) * 3;
    const float* ff = feat + ((size_t)b * Np + p) * C;
    h16* o = out + r * Kp;
    o[0] = sgn(pp[0] - cc[0]);
    o[1] = sgn(pp[1] - cc[1]);
    o[2] = sgn(pp[2] - cc[2]);
    for (int c = 0; c < C; ++c) o[3 + c] = sgn(ff[c]);
    for (int c = 3 + C; c < Kp; ++c) o[c] = (h16)0.0f;
  }
}

__global__ void group3(const float* __restrict__ posg, const float* __restrict__ feat,
                       int B, int S, int C, h16* __restrict__ out, int Kp)
{
  size_t rows = (size_t)B * S;
  for (size_t r = blockIdx.x * (size_t)blockDim.x + threadIdx.x; r < rows;
       r += (size_t)gridDim.x * blockDim.x) {
    const float* pp = posg + r * 3;
    const float* ff = feat + r * C;
    h16* o = out + r * Kp;
    o[0] = sgn(pp[0]); o[1] = sgn(pp[1]); o[2] = sgn(pp[2]);
    for (int c = 0; c < C; ++c) o[3 + c] = sgn(ff[c]);
    for (int c = 3 + C; c < Kp; ++c) o[c] = (h16)0.0f;
  }
}

// ============================================================================
// Batchnorm (batch statistics, population var) + ReLU, and binarize
// ============================================================================
__global__ void zero_f32(float* __restrict__ p, int n) {
  int i = blockIdx.x * blockDim.x + threadIdx.x;
  if (i < n) p[i] = 0.f;
}

__global__ void bn_stats(const float* __restrict__ X, size_t total, int N,
                         float* __restrict__ sums, float* __restrict__ sumsq)
{
  extern __shared__ float sdyn[];   // 2*N floats
  for (int i = threadIdx.x; i < 2 * N; i += blockDim.x) sdyn[i] = 0.f;
  __syncthreads();
  for (size_t idx = blockIdx.x * (size_t)blockDim.x + threadIdx.x; idx < total;
       idx += (size_t)gridDim.x * blockDim.x) {
    float v = X[idx];
    int n = (int)(idx % (size_t)N);
    atomicAdd(&sdyn[n], v);
    atomicAdd(&sdyn[N + n], v * v);
  }
  __syncthreads();
  for (int i = threadIdx.x; i < N; i += blockDim.x) {
    atomicAdd(&sums[i], sdyn[i]);
    atomicAdd(&sumsq[i], sdyn[N + i]);
  }
}

__global__ void bn_apply(float* __restrict__ X, size_t total, int N, int Mrows,
                         const float* __restrict__ sums, const float* __restrict__ sumsq,
                         const float* __restrict__ gamma, const float* __restrict__ beta)
{
  for (size_t idx = blockIdx.x * (size_t)blockDim.x + threadIdx.x; idx < total;
       idx += (size_t)gridDim.x * blockDim.x) {
    int n = (int)(idx % (size_t)N);
    float mu = sums[n] / (float)Mrows;
    float var = sumsq[n] / (float)Mrows - mu * mu;
    float y = (X[idx] - mu) * rsqrtf(var + BN_EPS_F) * gamma[n] + beta[n];
    X[idx] = fmaxf(y, 0.f);
  }
}

__global__ void binarize_pad(const float* __restrict__ X, size_t Mrows, int N,
                             h16* __restrict__ S, int Kp)
{
  size_t total = Mrows * (size_t)Kp;
  for (size_t idx = blockIdx.x * (size_t)blockDim.x + threadIdx.x; idx < total;
       idx += (size_t)gridDim.x * blockDim.x) {
    size_t row = idx / Kp; int c = (int)(idx % (size_t)Kp);
    h16 v = (h16)0.0f;
    if (c < N) v = sgn(X[row * N + c]);
    S[idx] = v;
  }
}

// ============================================================================
// Max-pool over middle axis: X (G, Kc, N) -> Out (G, N)
// ============================================================================
__global__ void maxpool_k(const float* __restrict__ X, int G, int Kc, int N,
                          float* __restrict__ Outp)
{
  size_t total = (size_t)G * N;
  for (size_t idx = blockIdx.x * (size_t)blockDim.x + threadIdx.x; idx < total;
       idx += (size_t)gridDim.x * blockDim.x) {
    int n = (int)(idx % (size_t)N);
    size_t g = idx / N;
    float m = -3.4e38f;
    for (int k = 0; k < Kc; ++k) m = fmaxf(m, X[(g * Kc + k) * (size_t)N + n]);
    Outp[idx] = m;
  }
}

__global__ void copy_out(const float* __restrict__ X, int M, int Npad, int Nl,
                         float* __restrict__ dst)
{
  int idx = blockIdx.x * blockDim.x + threadIdx.x;
  if (idx < M * Nl) dst[idx] = X[(size_t)(idx / Nl) * Npad + (idx % Nl)];
}

// ============================================================================
// Host orchestration
// ============================================================================
extern "C" void kernel_launch(void* const* d_in, const int* in_sizes, int n_in,
                              void* d_out, int out_size, void* d_ws, size_t ws_size,
                              hipStream_t stream)
{
  (void)in_sizes; (void)n_in; (void)out_size; (void)ws_size;
  const float* x = (const float*)d_in[0];   // (32, 4096, 3)

  // d_in flatten order: x; sa1 (W,g,b)x3; sa2 x3; sa3 x3; head W1,g1,b1,W2,g2,b2,Wo
  struct LDef { int wi, gidx, bidx; int O, I, Kp, Opad; };
  const LDef L[12] = {
    { 1,  2,  3,   64,    3,   32,   64},
    { 4,  5,  6,   64,   64,   64,   64},
    { 7,  8,  9,  128,   64,   64,  128},
    {10, 11, 12,  128,  131,  160,  128},
    {13, 14, 15,  128,  128,  128,  128},
    {16, 17, 18,  256,  128,  128,  256},
    {19, 20, 21,  256,  259,  288,  256},
    {22, 23, 24,  512,  256,  256,  512},
    {25, 26, 27, 1024,  512,  512, 1024},
    {28, 29, 30,  512, 1024, 1024,  512},
    {31, 32, 33,  256,  512,  512,  256},
    {34, -1, -1,   40,  256,  256,   48},
  };

  const int B = 32, N1 = 4096, S1 = 512, S2 = 128, KQ = 64;
  const size_t M1 = (size_t)B * S1 * KQ;   // 1,048,576
  const size_t M2 = (size_t)B * S2 * KQ;   // 262,144
  const size_t M3 = (size_t)B * S2;        // 4096
  const size_t MH = (size_t)B;             // 32

  uintptr_t basep = (uintptr_t)d_ws;
  size_t off = 0;
  auto carve = [&](size_t bytes) -> void* {
    off = (off + 255) & ~(size_t)255;
    void* p = (void*)(basep + off);
    off += bytes;
    return p;
  };

  float* bigf  = (float*)carve(M1 * 128 * sizeof(float));        // 512 MB (max pre-BN)
  h16*   signs = (h16*)  carve(M1 * 64 * sizeof(h16));           // 128 MB (max sign input)
  int*   cidx1 = (int*)  carve((size_t)B * S1 * sizeof(int));
  float* ctr1  = (float*)carve((size_t)B * S1 * 3 * sizeof(float));
  int*   gi1   = (int*)  carve(M1 * sizeof(int));
  int*   cidx2 = (int*)  carve((size_t)B * S2 * sizeof(int));
  float* ctr2  = (float*)carve((size_t)B * S2 * 3 * sizeof(float));
  int*   gi2   = (int*)  carve(M2 * sizeof(int));
  float* feat1 = (float*)carve((size_t)B * S1 * 128 * sizeof(float));
  float* feat2 = (float*)carve((size_t)B * S2 * 256 * sizeof(float));
  float* feat3 = (float*)carve((size_t)B * 1024 * sizeof(float));
  float* bnsum = (float*)carve(1024 * sizeof(float));
  float* bnsq  = (float*)carve(1024 * sizeof(float));
  float* outhd = (float*)carve((size_t)B * 48 * sizeof(float));
  h16* wsg[12]; float* wal[12];
  for (int i = 0; i < 12; ++i) {
    wsg[i] = (h16*)carve((size_t)L[i].Opad * L[i].Kp * sizeof(h16));
    wal[i] = (float*)carve((size_t)L[i].Opad * sizeof(float));
  }

  auto gsb = [](size_t total) -> int {
    size_t b = (total + 255) / 256;
    if (b > 16384) b = 16384;
    if (b < 1) b = 1;
    return (int)b;
  };

  // binarize weights + per-channel alpha (recomputed each call: deterministic)
  for (int i = 0; i < 12; ++i)
    prep_weights<<<L[i].Opad, 256, 0, stream>>>((const float*)d_in[L[i].wi],
                                                L[i].O, L[i].I, wsg[i], wal[i],
                                                L[i].Opad, L[i].Kp);

  auto run_gemm = [&](const h16* in, size_t M, int li, float* outp) {
    int tiles = (int)(M / 16) * (L[li].Opad / 16);
    int blocks = (tiles + 7) / 8;   // 8 waves per 256-thread block
    gemm_bin16<<<blocks, 256, 0, stream>>>(in, wsg[li], wal[li], outp,
                                           (int)M, L[li].Opad, L[li].Kp);
  };
  auto run_bn = [&](float* outp, size_t M, int li) {
    int Nc = L[li].Opad;
    size_t total = M * (size_t)Nc;
    zero_f32<<<(Nc + 255) / 256, 256, 0, stream>>>(bnsum, Nc);
    zero_f32<<<(Nc + 255) / 256, 256, 0, stream>>>(bnsq, Nc);
    bn_stats<<<gsb(total), 256, (size_t)2 * Nc * sizeof(float), stream>>>(
        outp, total, Nc, bnsum, bnsq);
    bn_apply<<<gsb(total), 256, 0, stream>>>(outp, total, Nc, (int)M, bnsum, bnsq,
                                             (const float*)d_in[L[li].gidx],
                                             (const float*)d_in[L[li].bidx]);
  };
  auto run_bin = [&](const float* src, size_t M, int Nc, int Kp) {
    binarize_pad<<<gsb(M * (size_t)Kp), 256, 0, stream>>>(src, M, Nc, signs, Kp);
  };

  // ---------------- SA stage 1 ----------------
  fps_kernel<<<B, 256, (size_t)4 * N1 * sizeof(float), stream>>>(x, B, N1, S1, cidx1, ctr1);
  ball_query_kernel<<<(B * S1 + 7) / 8, 256, 0, stream>>>(x, ctr1, B, N1, S1,
                                                          0.2f * 0.2f, KQ, gi1);
  group1<<<gsb(M1), 256, 0, stream>>>(x, ctr1, gi1, B, N1, S1, KQ, signs, 32);
  run_gemm(signs, M1, 0, bigf); run_bn(bigf, M1, 0);
  run_bin(bigf, M1, 64, 64);
  run_gemm(signs, M1, 1, bigf); run_bn(bigf, M1, 1);
  run_bin(bigf, M1, 64, 64);
  run_gemm(signs, M1, 2, bigf); run_bn(bigf, M1, 2);
  maxpool_k<<<gsb((size_t)B * S1 * 128), 256, 0, stream>>>(bigf, B * S1, KQ, 128, feat1);

  // ---------------- SA stage 2 ----------------
  fps_kernel<<<B, 256, (size_t)4 * S1 * sizeof(float), stream>>>(ctr1, B, S1, S2, cidx2, ctr2);
  ball_query_kernel<<<(B * S2 + 7) / 8, 256, 0, stream>>>(ctr1, ctr2, B, S1, S2,
                                                          0.4f * 0.4f, KQ, gi2);
  group2<<<gsb(M2), 256, 0, stream>>>(ctr1, ctr2, gi2, feat1, B, S1, S2, KQ, 128,
                                      signs, 160);
  run_gemm(signs, M2, 3, bigf); run_bn(bigf, M2, 3);
  run_bin(bigf, M2, 128, 128);
  run_gemm(signs, M2, 4, bigf); run_bn(bigf, M2, 4);
  run_bin(bigf, M2, 128, 128);
  run_gemm(signs, M2, 5, bigf); run_bn(bigf, M2, 5);
  maxpool_k<<<gsb((size_t)B * S2 * 256), 256, 0, stream>>>(bigf, B * S2, KQ, 256, feat2);

  // ---------------- global MLP (stage 3) ----------------
  group3<<<gsb(M3), 256, 0, stream>>>(ctr2, feat2, B, S2, 256, signs, 288);
  run_gemm(signs, M3, 6, bigf); run_bn(bigf, M3, 6);
  run_bin(bigf, M3, 256, 256);
  run_gemm(signs, M3, 7, bigf); run_bn(bigf, M3, 7);
  run_bin(bigf, M3, 512, 512);
  run_gemm(signs, M3, 8, bigf); run_bn(bigf, M3, 8);
  maxpool_k<<<gsb((size_t)B * 1024), 256, 0, stream>>>(bigf, B, S2, 1024, feat3);

  // ---------------- classifier head ----------------
  run_bin(feat3, MH, 1024, 1024);
  run_gemm(signs, MH, 9, bigf); run_bn(bigf, MH, 9);
  run_bin(bigf, MH, 512, 512);
  run_gemm(signs, MH, 10, bigf); run_bn(bigf, MH, 10);
  run_bin(bigf, MH, 256, 256);
  run_gemm(signs, MH, 11, outhd);                       // N padded to 48
  copy_out<<<(B * 40 + 255) / 256, 256, 0, stream>>>(outhd, B, 48, 40, (float*)d_out);
}
